// encoder_43516608643465
// MI455X (gfx1250) — matrix operand order repaired and checked by reading, not compiled, not run on hardware
//
#include <hip/hip_runtime.h>
#include <hip/hip_bf16.h>

// ---------------------------------------------------------------------------
// Types matching the gfx1250 WMMA builtin signatures
// ---------------------------------------------------------------------------
typedef __attribute__((ext_vector_type(16))) __bf16 v16bf;
typedef __attribute__((ext_vector_type(8)))  __bf16 v8bf;
typedef __attribute__((ext_vector_type(8)))  float  v8f;

#define S_SEL 9

// ---------------------------------------------------------------------------
// conv1: out0 = x @ W1 + b1   (x:[n,3], W1:[3,3])  -- tiny, VALU
// ---------------------------------------------------------------------------
__global__ void conv1_kernel(const float* __restrict__ x,
                             const float* __restrict__ W1,
                             const float* __restrict__ b1,
                             float* __restrict__ out0, int n)
{
    int i = blockIdx.x * blockDim.x + threadIdx.x;
    if (i >= n) return;
    float x0 = x[i * 3 + 0], x1 = x[i * 3 + 1], x2 = x[i * 3 + 2];
    #pragma unroll
    for (int co = 0; co < 3; ++co) {
        out0[i * 3 + co] = x0 * W1[0 * 3 + co] + x1 * W1[1 * 3 + co] +
                           x2 * W1[2 * 3 + co] + b1[co];
    }
}

// ---------------------------------------------------------------------------
// First sel-conv: Cin=3 (K=27), Cout=64, relu.  4 nodes per 256-thread block.
// ---------------------------------------------------------------------------
__global__ __launch_bounds__(256)
void selconv3_kernel(const float* __restrict__ xin,
                     const int*   __restrict__ src,
                     const float* __restrict__ interp,
                     const float* __restrict__ W,     // [9,3,64] row-major
                     const float* __restrict__ bias,
                     float* __restrict__ out)
{
    __shared__ float agg[4][28];
    const int g    = threadIdx.x >> 6;     // node group within block
    const int co   = threadIdx.x & 63;
    const int node = blockIdx.x * 4 + g;
    if (co < 27) {
        const int s = co / 3, ci = co - s * 3;
        const int e = node * S_SEL + s;
        agg[g][co] = interp[e] * xin[(long)src[e] * 3 + ci];
    }
    __syncthreads();
    float sum = bias[co];
    #pragma unroll
    for (int j = 0; j < 27; ++j) sum += agg[g][j] * W[j * 64 + co];
    out[(long)node * 64 + co] = sum > 0.f ? sum : 0.f;
}

// ---------------------------------------------------------------------------
// Weight pack: fp32 W[9,Cin,Cout] -> bf16 B-fragments.
// Fragment (kchunk c, ntile nt): 32 lanes x 16 bf16, 32x16 bf16 B, contiguous k order (same as A reads):
//   lane l: column n = nt*16 + (l&15); K = c*32 + ((l>>4)<<4) + e, e=0..15
// ---------------------------------------------------------------------------
__global__ void pack_w_kernel(const float* __restrict__ W,
                              __bf16* __restrict__ dst, int Cin, int Cout)
{
    const int g     = blockIdx.x * blockDim.x + threadIdx.x;
    const int NT    = Cout >> 4;
    const int nfrag = ((S_SEL * Cin) >> 5) * NT;
    const int frag  = g >> 5;
    const int lane  = g & 31;
    if (frag >= nfrag) return;
    const int c   = frag / NT;
    const int nt  = frag - c * NT;
    const int kg  = c * 32;              // flattened K = s*Cin + ci
    const int s   = kg / Cin;            // Cin % 32 == 0 -> chunk within one s
    const int ci0 = kg - s * Cin;
    const int n   = nt * 16 + (lane & 15);
    const int kb  = ci0 + ((lane >> 4) << 4);
    __bf16* o = dst + ((long)frag * 32 + lane) * 16;
    #pragma unroll
    for (int e = 0; e < 16; ++e)
        o[e] = (__bf16)W[((long)(s * Cin + kb + e)) * Cout + n];
}

// ---------------------------------------------------------------------------
// WMMA sel-conv: out[n,Cout] = relu( Ahat[n, 9*Cin] @ What + bias )
//   Ahat[n, s*Cin+ci] = interp[n*9+s] * xin[src[n*9+s], ci]   (bf16 staged)
//
// 128 threads = 4 waves. Block covers BM rows; waves partition N (NTW tiles
// each). M is register-blocked (MT = BM/16): each B fragment is loaded once
// per block per k-chunk and applied to MT A fragments -> weight L2 traffic
// divided by MT vs. a 16-row block.
// ---------------------------------------------------------------------------
template<int CIN, int COUT, int BM>
__global__ __launch_bounds__(128)
void selconv_wmma_kernel(const float* __restrict__ xin,
                         const int*   __restrict__ src,
                         const float* __restrict__ interp,
                         const __bf16* __restrict__ Wp,   // packed B fragments
                         const float* __restrict__ bias,
                         float* __restrict__ out)
{
    static_assert(CIN % 32 == 0 && COUT % 64 == 0 && BM % 16 == 0, "shape");
    constexpr int NT  = COUT / 16;   // total 16-col tiles
    constexpr int NTW = NT / 4;      // tiles per wave
    constexpr int KC  = CIN / 32;    // 32-K chunks per selection
    constexpr int MT  = BM / 16;     // 16-row tiles per block (register-blocked)
    constexpr int RPW = BM / 4;      // staging rows per wave
    constexpr int PAD = 8;           // 16B row pad: rotates LDS banks

    __shared__ alignas(16) __bf16 Abuf[BM][CIN + PAD];

    const int tid  = threadIdx.x;
    const int wave = tid >> 5;
    const int lane = tid & 31;
    const int m0   = blockIdx.x * BM;
    const int mrow = lane & 15;      // A row / D column this lane owns
    const int hi   = lane >> 4;

    v8f acc[MT][NTW] = {};

    for (int s = 0; s < S_SEL; ++s) {
        // ---- stage A_s = interp * gather(xin) into LDS as bf16 (paired) ----
        #pragma unroll
        for (int rr = 0; rr < RPW; ++rr) {
            const int r = wave * RPW + rr;
            const int e = (m0 + r) * S_SEL + s;
            const int   sn = src[e];
            const float w  = interp[e];
            const float* xr = xin + (long)sn * CIN;
            unsigned* drow = (unsigned*)&Abuf[r][0];
            for (int ci2 = lane; ci2 < CIN / 2; ci2 += 32) {
                union { __bf16 h[2]; unsigned u; } p;
                p.h[0] = (__bf16)(xr[ci2 * 2 + 0] * w);
                p.h[1] = (__bf16)(xr[ci2 * 2 + 1] * w);
                drow[ci2] = p.u;
            }
        }
        __syncthreads();

        #pragma unroll
        for (int c = 0; c < KC; ++c) {
            // A fragments (contiguous k = 16h + s, same order as packed B):
            //   lane<16: K 0..15 ; lane>=16: K 16..31
            const int klo = hi * 16;
            v16bf a[MT];
            #pragma unroll
            for (int mt = 0; mt < MT; ++mt) {
                union { v16bf v; v8bf h[2]; } u;
                u.h[0] = *(const v8bf*)&Abuf[mt * 16 + mrow][c * 32 + klo];
                u.h[1] = *(const v8bf*)&Abuf[mt * 16 + mrow][c * 32 + klo + 8];
                a[mt] = u.v;
            }
            const long kchunk = (long)s * KC + c;
            #pragma unroll
            for (int t = 0; t < NTW; ++t) {
                const int nt = wave * NTW + t;
                const v16bf b =
                    *(const v16bf*)&Wp[((kchunk * NT + nt) * 32 + lane) * 16];
                #pragma unroll
                for (int mt = 0; mt < MT; ++mt)
                    acc[mt][t] = __builtin_amdgcn_wmma_f32_16x16x32_bf16(
                        false, a[mt], false, b, (short)0, acc[mt][t], false, false);
            }
        }
        __syncthreads();
    }

    // ---- epilogue: bias + relu + store (ISA f32 16x16 C/D layout) ----
    #pragma unroll
    for (int t = 0; t < NTW; ++t) {
        const int col = (wave * NTW + t) * 16 + mrow;
        const float bv = bias[col];
        #pragma unroll
        for (int mt = 0; mt < MT; ++mt) {
            #pragma unroll
            for (int v = 0; v < 8; ++v) {
                const int row = m0 + mt * 16 + v + hi * 8;
                float val = acc[mt][t][v] + bv;
                out[(long)row * COUT + col] = val > 0.f ? val : 0.f;
            }
        }
    }
}

// ---------------------------------------------------------------------------
// segment_max with clus = i//4  ->  max over 4 consecutive rows
// ---------------------------------------------------------------------------
__global__ void pool4_kernel(const float* __restrict__ in,
                             float* __restrict__ out, int nout, int C)
{
    int g = blockIdx.x * blockDim.x + threadIdx.x;
    if (g >= nout * C) return;
    int j = g / C, c = g - j * C;
    const float* p = in + (long)(4 * j) * C + c;
    float m = p[0];
    m = fmaxf(m, p[C]);
    m = fmaxf(m, p[2 * C]);
    m = fmaxf(m, p[3 * C]);
    out[g] = m;
}

// ---------------------------------------------------------------------------
extern "C" void kernel_launch(void* const* d_in, const int* in_sizes, int n_in,
                              void* d_out, int out_size, void* d_ws, size_t ws_size,
                              hipStream_t stream)
{
    constexpr int N0 = 65536, N1 = 16384, N2 = 4096, N3 = 1024;

    const float* x       = (const float*)d_in[0];
    const int*   src0    = (const int*)  d_in[1];
    const float* interp0 = (const float*)d_in[4];
    const int*   src1    = (const int*)  d_in[5];
    const float* interp1 = (const float*)d_in[8];
    const int*   src2    = (const int*)  d_in[9];
    const float* interp2 = (const float*)d_in[12];
    const int*   src3    = (const int*)  d_in[13];
    const float* interp3 = (const float*)d_in[16];
    // d_in[17..19] = clus0..2 (implicit i//4, unused)
    const float* W1  = (const float*)d_in[20]; const float* b1  = (const float*)d_in[21];
    const float* W2  = (const float*)d_in[22]; const float* b2  = (const float*)d_in[23];
    const float* W3  = (const float*)d_in[24]; const float* b3  = (const float*)d_in[25];
    const float* W4  = (const float*)d_in[26]; const float* b4  = (const float*)d_in[27];
    const float* W5  = (const float*)d_in[28]; const float* b5  = (const float*)d_in[29];
    const float* W6  = (const float*)d_in[30]; const float* b6  = (const float*)d_in[31];
    const float* W7  = (const float*)d_in[32]; const float* b7  = (const float*)d_in[33];
    const float* W8  = (const float*)d_in[34]; const float* b8  = (const float*)d_in[35];
    const float* W9  = (const float*)d_in[36]; const float* b9  = (const float*)d_in[37];
    const float* W10 = (const float*)d_in[38]; const float* b10 = (const float*)d_in[39];

    // ---- workspace layout (256B aligned) ----
    char* ws = (char*)d_ws;
    float*  out0 = (float*)(ws + 0);            // N0*3*4   = 786432 B
    __bf16* pw3  = (__bf16*)(ws +  786432);     // 9*64*64*2
    __bf16* pw4  = (__bf16*)(ws +  860160);     // 9*64*128*2
    __bf16* pw5  = (__bf16*)(ws + 1007616);     // 9*128*128*2
    __bf16* pw6  = (__bf16*)(ws + 1302528);     // 9*128*256*2
    __bf16* pw7  = (__bf16*)(ws + 1892352);     // 9*256*256*2
    __bf16* pw8  = (__bf16*)(ws + 3072000);
    __bf16* pw9  = (__bf16*)(ws + 4251648);
    __bf16* pw10 = (__bf16*)(ws + 5431296);     // 9*256*512*2 -> end ~7.43MB

    // ---- d_out slices (fp32, concatenated in return order) ----
    float* o   = (float*)d_out;
    float* r11 = o;                 // N0*64
    float* r12 = o +  4194304;      // N0*64
    float* p1  = o +  8388608;      // N1*64
    float* r21 = o +  9437184;      // N1*128
    float* r22 = o + 11534336;      // N1*128
    float* p2  = o + 13631488;      // N2*128
    float* r31 = o + 14155776;      // N2*256
    float* r32 = o + 15204352;
    float* r33 = o + 16252928;
    float* r34 = o + 17301504;
    float* p3  = o + 18350080;      // N3*256
    float* r41 = o + 18612224;      // N3*512

    // ---- pack all WMMA weights (independent of data flow) ----
    auto pack = [&](const float* W, __bf16* dst, int Cin, int Cout) {
        int frags = ((S_SEL * Cin) >> 5) * (Cout >> 4);
        int thr = frags * 32;
        pack_w_kernel<<<(thr + 255) / 256, 256, 0, stream>>>(W, dst, Cin, Cout);
    };
    pack(W3,  pw3,   64,  64);
    pack(W4,  pw4,   64, 128);
    pack(W5,  pw5,  128, 128);
    pack(W6,  pw6,  128, 256);
    pack(W7,  pw7,  256, 256);
    pack(W8,  pw8,  256, 256);
    pack(W9,  pw9,  256, 256);
    pack(W10, pw10, 256, 512);

    // ---- network ----
    conv1_kernel<<<N0 / 256, 256, 0, stream>>>(x, W1, b1, out0, N0);
    selconv3_kernel<<<N0 / 4, 256, 0, stream>>>(out0, src0, interp0, W2, b2, r11);

    selconv_wmma_kernel< 64,  64, 64><<<N0 / 64, 128, 0, stream>>>(r11, src0, interp0, pw3, b3, r12);
    pool4_kernel<<<(N1 * 64 + 255) / 256, 256, 0, stream>>>(r12, p1, N1, 64);

    selconv_wmma_kernel< 64, 128, 64><<<N1 / 64, 128, 0, stream>>>(p1,  src1, interp1, pw4, b4, r21);
    selconv_wmma_kernel<128, 128, 64><<<N1 / 64, 128, 0, stream>>>(r21, src1, interp1, pw5, b5, r22);
    pool4_kernel<<<(N2 * 128 + 255) / 256, 256, 0, stream>>>(r22, p2, N2, 128);

    selconv_wmma_kernel<128, 256, 32><<<N2 / 32, 128, 0, stream>>>(p2,  src2, interp2, pw6, b6, r31);
    selconv_wmma_kernel<256, 256, 32><<<N2 / 32, 128, 0, stream>>>(r31, src2, interp2, pw7, b7, r32);
    selconv_wmma_kernel<256, 256, 32><<<N2 / 32, 128, 0, stream>>>(r32, src2, interp2, pw8, b8, r33);
    selconv_wmma_kernel<256, 256, 32><<<N2 / 32, 128, 0, stream>>>(r33, src2, interp2, pw9, b9, r34);
    pool4_kernel<<<(N3 * 256 + 255) / 256, 256, 0, stream>>>(r34, p3, N3, 256);

    selconv_wmma_kernel<256, 512, 32><<<N3 / 32, 128, 0, stream>>>(p3, src3, interp3, pw10, b10, r41);
}
